// TimeBERT_83554293777040
// MI455X (gfx1250) — matrix operand order repaired and checked by reading, not compiled, not run on hardware
//
#include <hip/hip_runtime.h>
#include <hip/hip_bf16.h>

typedef _Float16 v16h __attribute__((ext_vector_type(16)));
typedef float    v8f  __attribute__((ext_vector_type(8)));

#define WMMA_F16(a,b,c) __builtin_amdgcn_wmma_f32_16x16x32_f16(false,(a),false,(b),(short)0,(c),false,false)

// ---- WMMA fragment loaders (wave32; contiguous k order, same for A and B) ----
// A (16x32 f16): lane&15 = M row; hi = lane>>4 selects K half; element i -> K = k0 + 16*hi + i
__device__ __forceinline__ v16h ld_a(const _Float16* A, int lda, int row, int k0, int hi) {
  const _Float16* p = A + (size_t)row * lda + k0 + 16 * hi;
  v16h a;
#pragma unroll
  for (int i = 0; i < 8; ++i) { a[i] = p[i]; a[i + 8] = p[8 + i]; }
  return a;
}
// B (32x16 f16): lane&15 = N col; element i -> K = k0 + 16*hi + i (contiguous down column = row of W^T)
__device__ __forceinline__ v16h ld_b(const _Float16* Wt, int ldw, int col, int k0, int hi) {
  const _Float16* p = Wt + (size_t)col * ldw + k0 + 16 * hi;
  v16h b;
#pragma unroll
  for (int i = 0; i < 16; ++i) b[i] = p[i];
  return b;
}

// ---- Generic GEMM: C[M,N] = act(A[M,K] @ W[K,N] + bias); A f16 (lda=K), W given transposed f16 (N x K) ----
__global__ __launch_bounds__(128) void k_gemm(const _Float16* __restrict__ A,
                                              const _Float16* __restrict__ Wt,
                                              const float* __restrict__ bias,
                                              float* __restrict__ Cf, _Float16* __restrict__ Ch,
                                              int M, int N, int K, int act) {
  const int lane = threadIdx.x & 31, wave = threadIdx.x >> 5;
  const int n0 = (blockIdx.x * 4 + wave) * 16;
  if (n0 >= N) return;
  const int m0 = blockIdx.y * 16;
  const int hi = lane >> 4, nl = lane & 15;
  v8f acc = {};
  for (int k0 = 0; k0 < K; k0 += 32) {
    int ar = m0 + nl; if (ar > M - 1) ar = M - 1;
    v16h a = ld_a(A, K, ar, k0, hi);
    v16h b = ld_b(Wt, K, n0 + nl, k0, hi);
    acc = WMMA_F16(a, b, acc);
  }
  float bv = bias ? bias[n0 + nl] : 0.f;
#pragma unroll
  for (int r = 0; r < 8; ++r) {
    int row = m0 + r + 8 * hi;
    if (row < M) {
      float v = acc[r] + bv;
      if (act == 1) v = fmaxf(v, 0.f);
      else if (act == 2) v = tanhf(v);
      size_t o = (size_t)row * N + n0 + nl;
      if (Cf) Cf[o] = v;
      if (Ch) Ch[o] = (_Float16)v;
    }
  }
}

// ---- time embedding: [t*w_lin+b_lin, sin(t*w_per+b_per)] for B*L rows + 128 cls rows (t=i/127) ----
__global__ void k_time_emb(const float* __restrict__ ts, const float* __restrict__ w_per,
                           const float* __restrict__ b_per, const float* __restrict__ w_lin,
                           const float* __restrict__ b_lin, _Float16* __restrict__ key_e,
                           _Float16* __restrict__ cls_e, int nrows) {
  int idx = blockIdx.x * blockDim.x + threadIdx.x;
  int total = (nrows + 128) * 128;
  if (idx >= total) return;
  int row = idx >> 7, c = idx & 127;
  float t = (row < nrows) ? ts[row] : (float)(row - nrows) * (1.f / 127.f);
  float v = (c == 0) ? (t * w_lin[0] + b_lin[0]) : __sinf(t * w_per[c - 1] + b_per[c - 1]);
  if (row < nrows) key_e[(size_t)row * 128 + c] = (_Float16)v;
  else             cls_e[(size_t)(row - nrows) * 128 + c] = (_Float16)v;
}

// ---- f32 (RxC) -> f16 transposed (CxR) weight prep ----
__global__ void k_convT(const float* __restrict__ src, _Float16* __restrict__ dst, int R, int C) {
  int idx = blockIdx.x * blockDim.x + threadIdx.x;
  if (idx >= R * C) return;
  int r = idx / C, c = idx % C;
  dst[(size_t)c * R + r] = (_Float16)src[idx];
}

// ---- G^T (B,16,512): cols 0..7 = msk*vals, 8..15 = msk (num/den matrix for masked softmax-average) ----
__global__ void k_gt(const float* __restrict__ x, _Float16* __restrict__ Gt) {
  int idx = blockIdx.x * blockDim.x + threadIdx.x;
  if (idx >= 8 * 16 * 512) return;
  int b = idx >> 13, col = (idx >> 9) & 15, k = idx & 511;
  const float* xe = x + ((size_t)(b * 512 + k)) * 16;
  float v = (col < 8) ? xe[col] * xe[8 + col] : xe[8 + (col - 8)];
  Gt[idx] = (_Float16)v;
}

// ---- fused masked temporal attention ----
// per (qtile,h,b): S=QK^T/8 (WMMA, 8 waves) -> parallel rowmax -> E=exp -> R = E@G (WMMA, K split
// over 8 waves with LDS partial reduce) -> out = num/den
__global__ __launch_bounds__(256) void k_mta_attn(const _Float16* __restrict__ Qh,
                                                  const _Float16* __restrict__ Kh,
                                                  const _Float16* __restrict__ Gt,
                                                  _Float16* __restrict__ att, int Lq, int qstride) {
  __shared__ float S[16 * 512];
  __shared__ _Float16 E[16 * 512];
  __shared__ float Rp[8 * 256];   // per-wave partial 16x16 tiles
  __shared__ float R[256];
  __shared__ float red[256];
  __shared__ float rmax[16];
  const int b = blockIdx.z, h = blockIdx.y, qt = blockIdx.x;
  const int tid = threadIdx.x, lane = tid & 31, wave = tid >> 5;
  const int hi = lane >> 4, nl = lane & 15;
  const int rrow = tid >> 4, sub = tid & 15;   // 16 threads per row for reductions
  const _Float16* Qb = Qh + (size_t)b * qstride * 128 + h * 64;
  const _Float16* Kb = Kh + (size_t)b * 512 * 128 + h * 64;
  for (int nt = wave; nt < 32; nt += 8) {
    v8f acc = {};
#pragma unroll
    for (int k0 = 0; k0 < 64; k0 += 32) {
      v16h a  = ld_a(Qb, 128, qt * 16 + nl, k0, hi);
      v16h bb = ld_b(Kb, 128, nt * 16 + nl, k0, hi);
      acc = WMMA_F16(a, bb, acc);
    }
#pragma unroll
    for (int r = 0; r < 8; ++r) S[(r + 8 * hi) * 512 + nt * 16 + nl] = acc[r] * 0.125f;
  }
  __syncthreads();
  // parallel row max: 16 threads/row strided, then 4-step tree
  {
    float mx = -1e30f;
    for (int c = sub; c < 512; c += 16) mx = fmaxf(mx, S[rrow * 512 + c]);
    red[tid] = mx; __syncthreads();
#pragma unroll
    for (int s = 8; s > 0; s >>= 1) {
      if (sub < s) red[tid] = fmaxf(red[tid], red[tid + s]);
      __syncthreads();
    }
    if (sub == 0) rmax[rrow] = red[rrow * 16];
    __syncthreads();
  }
  for (int c = sub; c < 512; c += 16)
    E[rrow * 512 + c] = (_Float16)__expf(S[rrow * 512 + c] - rmax[rrow]);
  __syncthreads();
  // E(16x512) @ G(512x16): 16 K-chunks split 2-per-wave, partial tiles reduced in LDS
  {
    v8f acc = {};
#pragma unroll
    for (int kk = 0; kk < 2; ++kk) {
      int k0 = wave * 64 + kk * 32;
      v16h a  = ld_a(&E[0], 512, nl, k0, hi);
      v16h bb = ld_b(Gt + (size_t)b * 16 * 512, 512, nl, k0, hi);
      acc = WMMA_F16(a, bb, acc);
    }
#pragma unroll
    for (int r = 0; r < 8; ++r) Rp[wave * 256 + (r + 8 * hi) * 16 + nl] = acc[r];
  }
  __syncthreads();
  {
    float s = 0.f;
#pragma unroll
    for (int w = 0; w < 8; ++w) s += Rp[w * 256 + tid];
    R[tid] = s;
  }
  __syncthreads();
  {
    int r = tid >> 4, d = tid & 15;
    float den = R[r * 16 + (d & 7) + 8];
    float v = (den != 0.f) ? R[r * 16 + d] / den : 0.f;
    att[((size_t)b * Lq + qt * 16 + r) * 32 + h * 16 + d] = (_Float16)v;
  }
}

// ---- pad K to LPAD rows (zeros) and build V^T (B,H,64,LPAD) for WMMA B-operand ----
__global__ void k_repack(const _Float16* __restrict__ kh, const _Float16* __restrict__ vh,
                         _Float16* __restrict__ kpad, _Float16* __restrict__ vt, int Lq, int LPAD) {
  int idx = blockIdx.x * blockDim.x + threadIdx.x;
  int total = 8 * LPAD * 128;
  if (idx >= total) return;
  int c = idx & 127;
  int j = (idx >> 7) % LPAD;
  int b = idx / (LPAD * 128);
  _Float16 kv = (_Float16)0.f, vv = (_Float16)0.f;
  if (j < Lq) {
    size_t src = ((size_t)b * Lq + j) * 128 + c;
    kv = kh[src]; vv = vh[src];
  }
  kpad[((size_t)b * LPAD + j) * 128 + c] = kv;
  int hh = c >> 6, d = c & 63;
  vt[(((size_t)(b * 2 + hh)) * 64 + d) * LPAD + j] = vv;
}

// ---- fused transformer self-attention per (qtile,h,b) ----
// S in LDS (WMMA), parallel softmax over keys<Lq, O = (E@V)/rowsum with K split across wave pairs
#define EST 544
__global__ __launch_bounds__(256) void k_tb_attn(const _Float16* __restrict__ Qh,
                                                 const _Float16* __restrict__ Kpad,
                                                 const _Float16* __restrict__ Vt,
                                                 _Float16* __restrict__ Oh, int Lq, int LPAD) {
  __shared__ float S[16 * EST];
  __shared__ _Float16 E[16 * EST];
  __shared__ float Op[2 * 16 * 64];   // K-half partials of the 16x64 output tile
  __shared__ float red[256];
  __shared__ float rmax[16], rsum[16];
  const int b = blockIdx.z, h = blockIdx.y, qt = blockIdx.x;
  const int tid = threadIdx.x, lane = tid & 31, wave = tid >> 5;
  const int hi = lane >> 4, nl = lane & 15;
  const int rrow = tid >> 4, sub = tid & 15;
  const _Float16* Qb = Qh + (size_t)b * Lq * 128 + h * 64;
  const _Float16* Kb = Kpad + (size_t)b * LPAD * 128 + h * 64;
  const _Float16* Vb = Vt + ((size_t)(b * 2 + h)) * 64 * LPAD;
  const int ntiles = LPAD >> 4;
  for (int nt = wave; nt < ntiles; nt += 8) {
    v8f acc = {};
#pragma unroll
    for (int k0 = 0; k0 < 64; k0 += 32) {
      int qr = qt * 16 + nl; if (qr > Lq - 1) qr = Lq - 1;
      v16h a  = ld_a(Qb, 128, qr, k0, hi);
      v16h bb = ld_b(Kb, 128, nt * 16 + nl, k0, hi);
      acc = WMMA_F16(a, bb, acc);
    }
#pragma unroll
    for (int r = 0; r < 8; ++r) S[(r + 8 * hi) * EST + nt * 16 + nl] = acc[r] * 0.125f;
  }
  __syncthreads();
  // parallel row max over valid keys
  {
    float mx = -1e30f;
    for (int c = sub; c < Lq; c += 16) mx = fmaxf(mx, S[rrow * EST + c]);
    red[tid] = mx; __syncthreads();
#pragma unroll
    for (int s = 8; s > 0; s >>= 1) {
      if (sub < s) red[tid] = fmaxf(red[tid], red[tid + s]);
      __syncthreads();
    }
    if (sub == 0) rmax[rrow] = red[rrow * 16];
    __syncthreads();
  }
  // E = exp(S - rowmax) for c < Lq else 0; accumulate row sums in the same pass
  {
    float s = 0.f;
    for (int c = sub; c < LPAD; c += 16) {
      float e = (c < Lq) ? __expf(S[rrow * EST + c] - rmax[rrow]) : 0.f;
      E[rrow * EST + c] = (_Float16)e;
      s += e;
    }
    red[tid] = s; __syncthreads();
#pragma unroll
    for (int s2 = 8; s2 > 0; s2 >>= 1) {
      if (sub < s2) red[tid] += red[tid + s2];
      __syncthreads();
    }
    if (sub == 0) rsum[rrow] = red[rrow * 16];
    __syncthreads();
  }
  // O = E(16xLPAD) @ V(LPADx64): d-tile = wave&3, K-half = wave>>2; partial reduce in LDS
  {
    const int dt = wave & 3, khalf = wave >> 2;
    v8f acc = {};
    for (int k0 = khalf * 32; k0 < LPAD; k0 += 64) {
      v16h a  = ld_a(&E[0], EST, nl, k0, hi);
      v16h bb = ld_b(Vb, LPAD, dt * 16 + nl, k0, hi);
      acc = WMMA_F16(a, bb, acc);
    }
#pragma unroll
    for (int r = 0; r < 8; ++r)
      Op[(khalf * 16 + r + 8 * hi) * 64 + dt * 16 + nl] = acc[r];
  }
  __syncthreads();
  for (int i = tid; i < 16 * 64; i += 256) {
    int r = i >> 6, d = i & 63;
    int q = qt * 16 + r;
    if (q < Lq) {
      float v = (Op[r * 64 + d] + Op[(16 + r) * 64 + d]) / rsum[r];
      Oh[((size_t)b * Lq + q) * 128 + h * 64 + d] = (_Float16)v;
    }
  }
}

// ---- build transformer input: row0 = cls_emb, rows 1.. = src; optional +pos_emb; f32 + f16 copies ----
__global__ void k_assemble(const float* __restrict__ cls_emb, const float* __restrict__ src,
                           const float* __restrict__ pos, float* __restrict__ xf,
                           _Float16* __restrict__ xh, int Ls) {
  int Lt = Ls + 1;
  int idx = blockIdx.x * blockDim.x + threadIdx.x;
  int total = 8 * Lt * 128;
  if (idx >= total) return;
  int c = idx & 127;
  int i = (idx >> 7) % Lt;
  int b = idx / (Lt * 128);
  float v = (i == 0) ? cls_emb[c] : src[((size_t)b * Ls + i - 1) * 128 + c];
  if (pos) v += pos[(size_t)i * 128 + c];
  xf[idx] = v;
  xh[idx] = (_Float16)v;
}

// ---- residual add + LayerNorm over 128 features; one block per row ----
__global__ __launch_bounds__(128) void k_add_ln(const float* __restrict__ x, const float* __restrict__ r,
                                                const float* __restrict__ g, const float* __restrict__ bb,
                                                float* __restrict__ of, _Float16* __restrict__ oh) {
  __shared__ float red[128];
  int row = blockIdx.x, c = threadIdx.x;
  float v = x[(size_t)row * 128 + c] + r[(size_t)row * 128 + c];
  red[c] = v; __syncthreads();
  for (int s = 64; s > 0; s >>= 1) { if (c < s) red[c] += red[c + s]; __syncthreads(); }
  float mean = red[0] * (1.f / 128.f);
  __syncthreads();
  float d = v - mean;
  red[c] = d * d; __syncthreads();
  for (int s = 64; s > 0; s >>= 1) { if (c < s) red[c] += red[c + s]; __syncthreads(); }
  float var = red[0] * (1.f / 128.f);
  float y = d * rsqrtf(var + 1e-5f) * g[c] + bb[c];
  of[(size_t)row * 128 + c] = y;
  if (oh) oh[(size_t)row * 128 + c] = (_Float16)y;
}

__global__ void k_gather_cls(const float* __restrict__ hf, _Float16* __restrict__ clsA) {
  int idx = blockIdx.x * blockDim.x + threadIdx.x;
  if (idx >= 8 * 128) return;
  int b = idx >> 7, c = idx & 127;
  clsA[idx] = (_Float16)hf[((size_t)b * 129) * 128 + c];
}

__global__ void k_final(const float* __restrict__ h2f, float* __restrict__ out) {
  int idx = blockIdx.x * blockDim.x + threadIdx.x;
  if (idx >= 8 * 512 * 128) return;
  int c = idx & 127;
  int i = (idx >> 7) & 511;
  int b = idx / (512 * 128);
  out[idx] = h2f[((size_t)b * 513 + 1 + i) * 128 + c];
}

extern "C" void kernel_launch(void* const* d_in, const int* in_sizes, int n_in,
                              void* d_out, int out_size, void* d_ws, size_t ws_size,
                              hipStream_t stream) {
  (void)in_sizes; (void)n_in; (void)out_size; (void)ws_size;
  const float* x     = (const float*)d_in[0];
  const float* ts    = (const float*)d_in[1];
  const float* w_per = (const float*)d_in[2];
  const float* b_per = (const float*)d_in[3];
  const float* w_lin = (const float*)d_in[4];
  const float* b_lin = (const float*)d_in[5];
  const float* Wq_t  = (const float*)d_in[6];
  const float* bq_t  = (const float*)d_in[7];
  const float* Wk_t  = (const float*)d_in[8];
  const float* bk_t  = (const float*)d_in[9];
  const float* Wo_t  = (const float*)d_in[10];
  const float* bo_t  = (const float*)d_in[11];
  const float* pos   = (const float*)d_in[12];
  const float* clsE  = (const float*)d_in[13];
  const float* tWq = (const float*)d_in[14]; const float* tbq = (const float*)d_in[15];
  const float* tWk = (const float*)d_in[16]; const float* tbk = (const float*)d_in[17];
  const float* tWv = (const float*)d_in[18]; const float* tbv = (const float*)d_in[19];
  const float* tWo = (const float*)d_in[20]; const float* tbo = (const float*)d_in[21];
  const float* ln1g = (const float*)d_in[22]; const float* ln1b = (const float*)d_in[23];
  const float* fW1 = (const float*)d_in[24]; const float* fb1 = (const float*)d_in[25];
  const float* fW2 = (const float*)d_in[26]; const float* fb2 = (const float*)d_in[27];
  const float* ln2g = (const float*)d_in[28]; const float* ln2b = (const float*)d_in[29];
  const float* pW = (const float*)d_in[30]; const float* pb = (const float*)d_in[31];
  float* out = (float*)d_out;

  char* base = (char*)d_ws; size_t off = 0;
  auto alloc = [&](size_t elems, size_t esz) -> void* {
    void* p = base + off;
    off += (elems * esz + 255) & ~(size_t)255;
    return p;
  };
  auto ah = [&](size_t e) { return (_Float16*)alloc(e, 2); };
  auto af = [&](size_t e) { return (float*)alloc(e, 4); };

  _Float16 *wqT = ah(16384), *wkT = ah(16384), *woT = ah(4096);
  _Float16 *twqT = ah(16384), *twkT = ah(16384), *twvT = ah(16384), *twoT = ah(16384);
  _Float16 *fw1T = ah(16384), *fw2T = ah(16384), *pwT = ah(16384);
  _Float16 *key_e = ah((size_t)4096 * 128), *cls_e = ah(128 * 128);
  _Float16 *kproj = ah((size_t)4096 * 128), *qproj = ah((size_t)4096 * 128), *clsq = ah(128 * 128);
  _Float16 *Gt = ah(8 * 16 * 512);
  _Float16 *attC = ah(8 * 128 * 32), *attO = ah((size_t)8 * 512 * 32);
  float *clsOut = af((size_t)8 * 128 * 128), *mtaOut = af((size_t)8 * 512 * 128);
  const int RMAX = 8 * 513;
  float *xf = af((size_t)RMAX * 128); _Float16 *xh = ah((size_t)RMAX * 128);
  _Float16 *qh = ah((size_t)RMAX * 128), *kh = ah((size_t)RMAX * 128), *vh = ah((size_t)RMAX * 128);
  _Float16 *kpad = ah((size_t)8 * 544 * 128), *vt = ah((size_t)8 * 2 * 64 * 544);
  _Float16 *oh = ah((size_t)RMAX * 128);
  float *tmpf = af((size_t)RMAX * 128);
  float *x1f = af((size_t)RMAX * 128); _Float16 *x1h = ah((size_t)RMAX * 128);
  _Float16 *midh = ah((size_t)RMAX * 128);
  float *outf = af((size_t)RMAX * 128);
  _Float16 *clsA = ah(8 * 128);

  auto cdiv = [](int a, int b) { return (a + b - 1) / b; };

  auto convT = [&](const float* s, _Float16* d, int R, int C) {
    k_convT<<<cdiv(R * C, 256), 256, 0, stream>>>(s, d, R, C);
  };
  convT(Wq_t, wqT, 128, 128); convT(Wk_t, wkT, 128, 128); convT(Wo_t, woT, 32, 128);
  convT(tWq, twqT, 128, 128); convT(tWk, twkT, 128, 128); convT(tWv, twvT, 128, 128);
  convT(tWo, twoT, 128, 128); convT(fW1, fw1T, 128, 128); convT(fW2, fw2T, 128, 128);
  convT(pW, pwT, 128, 128);

  k_time_emb<<<cdiv((4096 + 128) * 128, 256), 256, 0, stream>>>(ts, w_per, b_per, w_lin, b_lin,
                                                                key_e, cls_e, 4096);

  auto gemm = [&](const _Float16* A, const _Float16* Wt, const float* bias, float* Cf,
                  _Float16* Ch, int M, int N, int K, int act) {
    dim3 g(cdiv(N / 16, 4), cdiv(M, 16));
    k_gemm<<<g, 128, 0, stream>>>(A, Wt, bias, Cf, Ch, M, N, K, act);
  };

  // q/k projections (q_emb == key_e since tq_e == key_e)
  gemm(key_e, wkT, bk_t, nullptr, kproj, 4096, 128, 128, 0);
  gemm(key_e, wqT, bq_t, nullptr, qproj, 4096, 128, 128, 0);
  gemm(cls_e, wqT, bq_t, nullptr, clsq, 128, 128, 128, 0);

  k_gt<<<cdiv(8 * 16 * 512, 256), 256, 0, stream>>>(x, Gt);

  // masked temporal attention (cls queries broadcast over batch -> qstride 0)
  k_mta_attn<<<dim3(8, 2, 8), 256, 0, stream>>>(clsq, kproj, Gt, attC, 128, 0);
  k_mta_attn<<<dim3(32, 2, 8), 256, 0, stream>>>(qproj, kproj, Gt, attO, 512, 512);

  gemm(attC, woT, bo_t, clsOut, nullptr, 1024, 128, 32, 0);
  gemm(attO, woT, bo_t, mtaOut, nullptr, 4096, 128, 32, 0);

  auto tblock = [&](const float* srcRows, const float* posE, int Ls, float* finalF) {
    int Lq = Ls + 1, LPAD = ((Lq + 31) / 32) * 32;
    int Mrows = 8 * Lq;
    k_assemble<<<cdiv(Mrows * 128, 256), 256, 0, stream>>>(clsE, srcRows, posE, xf, xh, Ls);
    gemm(xh, twqT, tbq, nullptr, qh, Mrows, 128, 128, 0);
    gemm(xh, twkT, tbk, nullptr, kh, Mrows, 128, 128, 0);
    gemm(xh, twvT, tbv, nullptr, vh, Mrows, 128, 128, 0);
    k_repack<<<cdiv(8 * LPAD * 128, 256), 256, 0, stream>>>(kh, vh, kpad, vt, Lq, LPAD);
    k_tb_attn<<<dim3(cdiv(Lq, 16), 2, 8), 256, 0, stream>>>(qh, kpad, vt, oh, Lq, LPAD);
    gemm(oh, twoT, tbo, tmpf, nullptr, Mrows, 128, 128, 0);
    k_add_ln<<<Mrows, 128, 0, stream>>>(xf, tmpf, ln1g, ln1b, x1f, x1h);
    gemm(x1h, fw1T, fb1, nullptr, midh, Mrows, 128, 128, 1);
    gemm(midh, fw2T, fb2, tmpf, nullptr, Mrows, 128, 128, 0);
    k_add_ln<<<Mrows, 128, 0, stream>>>(x1f, tmpf, ln2g, ln2b, finalF, nullptr);
  };

  // cls branch: tblock on (B,129,128), then tanh pooling -> out[0:1024]
  tblock(clsOut, nullptr, 128, outf);
  k_gather_cls<<<cdiv(1024, 256), 256, 0, stream>>>(outf, clsA);
  gemm(clsA, pwT, pb, out, nullptr, 8, 128, 128, 2);

  // sequence branch: tblock on (B,513,128) with pos_emb, drop cls row -> out[1024:]
  tblock(mtaOut, pos, 512, outf);
  k_final<<<cdiv(8 * 512 * 128, 256), 256, 0, stream>>>(outf, out + 1024);
}
